// Qwen2DecoderLayer_11115375362902
// MI455X (gfx1250) — matrix operand order repaired and checked by reading, not compiled, not run on hardware
//
#include <hip/hip_runtime.h>
#include <math.h>

// ---------------------------------------------------------------------------
// Qwen2 decoder layer for gfx1250 (MI455X).  B=2, T=2048, D=2048, N=16 heads,
// KV=2 heads, H=128, I=5504.  All GEMMs in bf16 WMMA (v_wmma_f32_16x16x32_bf16)
// with f32 accumulation; softmax / rmsnorm / rope in f32.  B panels are staged
// into LDS with double-buffered global_load_async_to_lds_b128 (ASYNCcnt).
// ---------------------------------------------------------------------------

typedef __attribute__((ext_vector_type(16))) __bf16 v16bf;
typedef __attribute__((ext_vector_type(8)))  __bf16 v8bf;
typedef __attribute__((ext_vector_type(8)))  float  v8f;

union ABu { v16bf v; v8bf h[2]; };
union V8FU { v8f v; float f[8]; };

#define WMMA_BF16(a, b, c) \
    __builtin_amdgcn_wmma_f32_16x16x32_bf16(false, (a), false, (b), (short)0, (c), false, false)

// ---------------------------------------------------------------------------
// RMSNorm: one block per row of D=2048, f32 in -> bf16 out
// ---------------------------------------------------------------------------
__global__ __launch_bounds__(256) void rmsnorm_kernel(
    const float* __restrict__ x, const float* __restrict__ g,
    __bf16* __restrict__ out, int Dd)
{
  __shared__ float ssum[8];
  const int row = blockIdx.x;
  const float* xr = x + (size_t)row * Dd;
  float v[8];
  float acc = 0.f;
#pragma unroll
  for (int i = 0; i < 8; ++i) {
    v[i] = xr[threadIdx.x + i * 256];
    acc += v[i] * v[i];
  }
#pragma unroll
  for (int off = 16; off > 0; off >>= 1) acc += __shfl_xor(acc, off, 32);
  if ((threadIdx.x & 31) == 0) ssum[threadIdx.x >> 5] = acc;
  __syncthreads();
  float tot = 0.f;
#pragma unroll
  for (int i = 0; i < 8; ++i) tot += ssum[i];
  const float r = rsqrtf(tot / (float)Dd + 1e-6f);
#pragma unroll
  for (int i = 0; i < 8; ++i) {
    const int c = threadIdx.x + i * 256;
    out[(size_t)row * Dd + c] = (__bf16)(v[i] * r * g[c]);
  }
}

// ---------------------------------------------------------------------------
// Pack a row-major f32 [Kd x N] weight into WMMA B-fragment tile order:
// tiles (kt,nt) of 32x16, tile data lane-major: elem (lane,e) holds
// B[kt*32 + 16*(lane>>4) + e][nt*16 + (lane&15)].
// ---------------------------------------------------------------------------
__global__ __launch_bounds__(256) void pack_b_kernel(
    const float* __restrict__ src, __bf16* __restrict__ dst, int Kd, int N)
{
  const size_t idx = (size_t)blockIdx.x * 256 + threadIdx.x;
  const size_t total = (size_t)Kd * N;
  if (idx >= total) return;
  const int within = (int)(idx & 511);
  const size_t tile = idx >> 9;
  const int lane = within >> 4, e = within & 15;
  const int NT = N >> 4;
  const int kt = (int)(tile / NT), nt = (int)(tile % NT);
  const int k = kt * 32 + ((lane >> 4) << 4) + e;
  const int n = nt * 16 + (lane & 15);
  dst[idx] = (__bf16)src[(size_t)k * N + n];
}

// Pack fused QKV weight: columns 0..2047 = q_w(N,D,H), 2048..2303 = k_w(K,D,H),
// 2304..2559 = v_w(K,D,H); contraction dim k is d.
__global__ __launch_bounds__(256) void pack_qkv_w_kernel(
    const float* __restrict__ qw, const float* __restrict__ kw,
    const float* __restrict__ vw, __bf16* __restrict__ dst)
{
  const size_t idx = (size_t)blockIdx.x * 256 + threadIdx.x;
  if (idx >= (size_t)2048 * 2560) return;
  const int within = (int)(idx & 511);
  const size_t tile = idx >> 9;
  const int lane = within >> 4, e = within & 15;
  const int NT = 2560 >> 4;  // 160
  const int kt = (int)(tile / NT), nt = (int)(tile % NT);
  const int d = kt * 32 + ((lane >> 4) << 4) + e;
  const int c = nt * 16 + (lane & 15);
  float val;
  if (c < 2048) {
    const int hn = c >> 7, hh = c & 127;
    val = qw[(size_t)hn * 2048 * 128 + (size_t)d * 128 + hh];
  } else if (c < 2304) {
    const int c2 = c - 2048, hn = c2 >> 7, hh = c2 & 127;
    val = kw[(size_t)hn * 2048 * 128 + (size_t)d * 128 + hh];
  } else {
    const int c2 = c - 2304, hn = c2 >> 7, hh = c2 & 127;
    val = vw[(size_t)hn * 2048 * 128 + (size_t)d * 128 + hh];
  }
  dst[idx] = (__bf16)val;
}

__global__ __launch_bounds__(256) void pack_qkv_bias_kernel(
    const float* __restrict__ qb, const float* __restrict__ kb,
    const float* __restrict__ vb, float* __restrict__ bias)
{
  const int idx = blockIdx.x * 256 + threadIdx.x;
  if (idx >= 2560) return;
  bias[idx] = (idx < 2048) ? qb[idx] : (idx < 2304) ? kb[idx - 2048] : vb[idx - 2304];
}

// ---------------------------------------------------------------------------
// Generic bf16 WMMA GEMM:  C[M x N] = A[M x Kd] * B(packed tiles)[Kd x N]
// Block = 256 threads = 8 waves arranged 4(M) x 2(N); wave owns a 32x32 tile;
// block tile = 128 x 64; grid (M/128, N/64).
// The block's 64-column B panel (4 KB) is staged into LDS via double-buffered
// global_load_async_to_lds_b128 and consumed with ds loads.
// Epilogue modes:
//   0: outf = acc
//   1: outf = acc + bias[col]
//   2: outf = acc + extra[row,col]          (residual add)
//   3: outb = silu(extra[row,col]) * acc    (bf16 out; extra = gate f32)
// ---------------------------------------------------------------------------
__global__ __launch_bounds__(256) void gemm_bf16_kernel(
    const __bf16* __restrict__ A, const __bf16* __restrict__ Bt,
    const float* __restrict__ bias, const float* __restrict__ extra,
    float* __restrict__ outf, __bf16* __restrict__ outb,
    int M, int N, int Kd, int mode)
{
  __shared__ __bf16 bstage[2][2048];  // two 4 KB panels (64 cols x 32 k)

  const int tid = threadIdx.x;
  const int lane = tid & 31;
  const int wave = tid >> 5;
  const int half = lane >> 4;
  const int l15 = lane & 15;
  const int mw = wave >> 1;  // 0..3
  const int nw = wave & 1;   // 0..1
  const int m0 = blockIdx.x * 128 + mw * 32;
  const int n0 = blockIdx.y * 64 + nw * 32;
  const int ktiles = Kd >> 5;
  const int NT = N >> 4;
  const int ntile0 = blockIdx.y * 4;  // first of 4 packed B tiles for this block

  const v8f vzero = {0.f, 0.f, 0.f, 0.f, 0.f, 0.f, 0.f, 0.f};
  v8f acc00 = vzero, acc01 = vzero, acc10 = vzero, acc11 = vzero;

  const __bf16* arow0 = A + (size_t)(m0 + l15) * Kd;
  const __bf16* arow1 = A + (size_t)(m0 + 16 + l15) * Kd;

  // Each thread DMAs 16 bytes of the panel: async global -> LDS (ASYNCcnt).
  auto issue_panel = [&](int s, int kt) {
    const __bf16* src = Bt + ((size_t)kt * NT + ntile0) * 512 + tid * 8;
    const unsigned dst = (unsigned)(size_t)(&bstage[s][tid * 8]);
    asm volatile("global_load_async_to_lds_b128 %0, %1, off"
                 :: "v"(dst), "v"((unsigned long long)(size_t)src)
                 : "memory");
  };

  issue_panel(0, 0);
  for (int kt = 0; kt < ktiles; ++kt) {
    if (kt + 1 < ktiles) {
      issue_panel((kt + 1) & 1, kt + 1);
      asm volatile("s_wait_asynccnt 0x1" ::: "memory");  // panel kt landed
    } else {
      asm volatile("s_wait_asynccnt 0x0" ::: "memory");
    }
    __syncthreads();  // panel kt visible to all waves

    const __bf16* bs = &bstage[kt & 1][0];
    ABu a0, a1;
    const __bf16* ap0 = arow0 + kt * 32 + 16 * half;
    const __bf16* ap1 = arow1 + kt * 32 + 16 * half;
    a0.h[0] = *(const v8bf*)ap0;
    a0.h[1] = *(const v8bf*)(ap0 + 8);
    a1.h[0] = *(const v8bf*)ap1;
    a1.h[1] = *(const v8bf*)(ap1 + 8);

    v16bf b0 = *(const v16bf*)(bs + (nw * 2 + 0) * 512 + lane * 16);
    v16bf b1 = *(const v16bf*)(bs + (nw * 2 + 1) * 512 + lane * 16);

    acc00 = WMMA_BF16(a0.v, b0, acc00);
    acc01 = WMMA_BF16(a0.v, b1, acc01);
    acc10 = WMMA_BF16(a1.v, b0, acc10);
    acc11 = WMMA_BF16(a1.v, b1, acc11);

    __syncthreads();  // everyone done with panel kt&1 before it is overwritten
  }

  auto emit = [&](int row, int col, float v) {
    const size_t i = (size_t)row * N + col;
    if (mode == 0) {
      outf[i] = v;
    } else if (mode == 1) {
      outf[i] = v + bias[col];
    } else if (mode == 2) {
      outf[i] = v + extra[i];
    } else {  // silu(gate) * up -> bf16
      const float gv = extra[i];
      const float sv = gv / (1.0f + expf(-gv));
      outb[i] = (__bf16)(sv * v);
    }
  };

  V8FU u00, u01, u10, u11;
  u00.v = acc00; u01.v = acc01; u10.v = acc10; u11.v = acc11;
  const int colA = n0 + l15;
  const int colB = n0 + 16 + l15;
#pragma unroll
  for (int r = 0; r < 8; ++r) {
    const int row0 = m0 + r + 8 * half;
    const int row1 = row0 + 16;
    emit(row0, colA, u00.f[r]);
    emit(row0, colB, u01.f[r]);
    emit(row1, colA, u10.f[r]);
    emit(row1, colB, u11.f[r]);
  }
}

// ---------------------------------------------------------------------------
// QKV post-processing: RoPE on q (with 1/sqrt(H) scale) and k, transpose v.
// qkv f32 layout: [B*T, 2560] with cols 0..2047 q, 2048..2303 k, 2304..2559 v.
// ---------------------------------------------------------------------------
__global__ __launch_bounds__(256) void rope_q_kernel(
    const float* __restrict__ qkv, const int* __restrict__ pos,
    __bf16* __restrict__ qo)
{
  const int idx = blockIdx.x * 256 + threadIdx.x;  // B*N*T*64 = 4194304
  const int i = idx & 63;
  const int t = (idx >> 6) & 2047;
  const int n = (idx >> 17) & 15;
  const int b = idx >> 21;
  const size_t base = ((size_t)(b * 2048 + t)) * 2560 + n * 128;
  const float q1 = qkv[base + i];
  const float q2 = qkv[base + i + 64];
  const float freq = expf(-(float)i * (13.815510557964274f / 64.0f));  // 1e6^(-i/64)
  const float ang = (float)pos[t] * freq;
  float s, c;
  sincosf(ang, &s, &c);
  const float scale = 0.08838834764831845f;  // 128^-0.5
  const size_t ob = ((size_t)((b * 16 + n) * 2048 + t)) * 128;
  qo[ob + i]      = (__bf16)((q1 * c - q2 * s) * scale);
  qo[ob + i + 64] = (__bf16)((q2 * c + q1 * s) * scale);
}

__global__ __launch_bounds__(256) void rope_k_kernel(
    const float* __restrict__ qkv, const int* __restrict__ pos,
    __bf16* __restrict__ ko)
{
  const int idx = blockIdx.x * 256 + threadIdx.x;  // B*KV*T*64 = 524288
  const int i = idx & 63;
  const int t = (idx >> 6) & 2047;
  const int kv = (idx >> 17) & 1;
  const int b = idx >> 18;
  const size_t base = ((size_t)(b * 2048 + t)) * 2560 + 2048 + kv * 128;
  const float k1 = qkv[base + i];
  const float k2 = qkv[base + i + 64];
  const float freq = expf(-(float)i * (13.815510557964274f / 64.0f));
  const float ang = (float)pos[t] * freq;
  float s, c;
  sincosf(ang, &s, &c);
  const size_t ob = ((size_t)((b * 2 + kv) * 2048 + t)) * 128;
  ko[ob + i]      = (__bf16)(k1 * c - k2 * s);
  ko[ob + i + 64] = (__bf16)(k2 * c + k1 * s);
}

__global__ __launch_bounds__(256) void vtrans_kernel(
    const float* __restrict__ qkv, __bf16* __restrict__ vt)
{
  const int idx = blockIdx.x * 256 + threadIdx.x;  // B*KV*T*H = 1048576
  const int h = idx & 127;
  const int t = (idx >> 7) & 2047;
  const int kv = (idx >> 18) & 1;
  const int b = idx >> 19;
  const float val = qkv[((size_t)(b * 2048 + t)) * 2560 + 2304 + kv * 128 + h];
  vt[((size_t)((b * 2 + kv) * 128 + h)) * 2048 + t] = (__bf16)val;
}

// ---------------------------------------------------------------------------
// Flash attention.  One wave per 16-query tile; online softmax over 32-key
// steps.  Q: [B,N,T,H] bf16 (pre-scaled), K: [B,KV,T,H] bf16,
// Vt: [B,KV,H,T] bf16.  O: [B*T, N*H] bf16 row-major (A matrix of o-proj).
// ---------------------------------------------------------------------------
__global__ __launch_bounds__(256) void attn_kernel(
    const __bf16* __restrict__ Q, const __bf16* __restrict__ Kb,
    const __bf16* __restrict__ Vt, __bf16* __restrict__ O)
{
  constexpr int T = 2048, H = 128, NHEAD = 16, KV = 2;
  __shared__ float pbuf[8][16 * 32];

  const int lane = threadIdx.x & 31;
  const int wave = threadIdx.x >> 5;
  const int half = lane >> 4;
  const int l15 = lane & 15;

  const int tile = blockIdx.x * 8 + wave;  // B*N*(T/16) = 4096 tiles
  const int qt = tile & 127;
  const int head = tile >> 7;
  const int n = head & 15;
  const int b = head >> 4;
  const int kv = n >> 3;  // N/KV = 8 heads per kv group
  const int t0 = qt << 4;

  // Resident Q fragments (16 x 128 over four K=32 steps)
  ABu qa[4];
  {
    const __bf16* qbase = Q + ((size_t)((b * NHEAD + n) * T + t0 + l15)) * H;
#pragma unroll
    for (int ks = 0; ks < 4; ++ks) {
      const __bf16* p = qbase + ks * 32 + 16 * half;
      qa[ks].h[0] = *(const v8bf*)p;
      qa[ks].h[1] = *(const v8bf*)(p + 8);
    }
  }
  const __bf16* kbase = Kb + ((size_t)(b * KV + kv) * T) * H;
  const __bf16* vbase = Vt + ((size_t)(b * KV + kv) * H) * T;

  const v8f vzero = {0.f, 0.f, 0.f, 0.f, 0.f, 0.f, 0.f, 0.f};
  v8f acc[8];
#pragma unroll
  for (int i = 0; i < 8; ++i) acc[i] = vzero;
  float mrow[8], lrow[8];
#pragma unroll
  for (int r = 0; r < 8; ++r) { mrow[r] = -__builtin_inff(); lrow[r] = 0.0f; }

  float* pb = &pbuf[wave][0];

  for (int s0 = 0; s0 <= t0; s0 += 32) {
    // ---- S = Q * K^T for 32 keys (two 16x16 column tiles) ----
    v8f s0acc = vzero, s1acc = vzero;
#pragma unroll
    for (int ks = 0; ks < 4; ++ks) {
      v16bf kb0 = *(const v16bf*)(kbase + (size_t)(s0 + l15) * H + ks * 32 + 16 * half);
      v16bf kb1 = *(const v16bf*)(kbase + (size_t)(s0 + 16 + l15) * H + ks * 32 + 16 * half);
      s0acc = WMMA_BF16(qa[ks].v, kb0, s0acc);
      s1acc = WMMA_BF16(qa[ks].v, kb1, s1acc);
    }

    // ---- causal mask + online softmax ----
    V8FU sa, sb;
    sa.v = s0acc; sb.v = s1acc;
    const int colA = s0 + l15;
    const int colB = s0 + 16 + l15;
    float scl[8];
#pragma unroll
    for (int r = 0; r < 8; ++r) {
      const int rowt = t0 + r + 8 * half;
      float a = (colA <= rowt) ? sa.f[r] : -__builtin_inff();
      float c = (colB <= rowt) ? sb.f[r] : -__builtin_inff();
      float mx = fmaxf(a, c);
      mx = fmaxf(mx, __shfl_xor(mx, 1, 32));
      mx = fmaxf(mx, __shfl_xor(mx, 2, 32));
      mx = fmaxf(mx, __shfl_xor(mx, 4, 32));
      mx = fmaxf(mx, __shfl_xor(mx, 8, 32));
      const float mn = fmaxf(mrow[r], mx);
      const float sc = __expf(mrow[r] - mn);
      const float pa = __expf(a - mn);
      const float pc = __expf(c - mn);
      float rs = pa + pc;
      rs += __shfl_xor(rs, 1, 32);
      rs += __shfl_xor(rs, 2, 32);
      rs += __shfl_xor(rs, 4, 32);
      rs += __shfl_xor(rs, 8, 32);
      lrow[r] = lrow[r] * sc + rs;
      mrow[r] = mn;
      scl[r] = sc;
      // stash P in D-fragment order for the LDS transpose
      pb[(r + 8 * half) * 32 + l15] = pa;
      pb[(r + 8 * half) * 32 + 16 + l15] = pc;
    }

    // ---- rescale accumulators ----
#pragma unroll
    for (int nt = 0; nt < 8; ++nt) {
      V8FU u; u.v = acc[nt];
#pragma unroll
      for (int r = 0; r < 8; ++r) u.f[r] *= scl[r];
      acc[nt] = u.v;
    }

    // ---- P: D-layout (LDS) -> A-fragment bf16 ----
    union { v16bf v; __bf16 e[16]; } pf;
#pragma unroll
    for (int e = 0; e < 16; ++e) {
      const int k = 16 * half + e;
      pf.e[e] = (__bf16)pb[l15 * 32 + k];
    }

    // ---- O += P * V (contraction over 32 keys, 8 h-tiles) ----
#pragma unroll
    for (int nt = 0; nt < 8; ++nt) {
      v16bf vb = *(const v16bf*)(vbase + (size_t)(nt * 16 + l15) * T + s0 + 16 * half);
      acc[nt] = WMMA_BF16(pf.v, vb, acc[nt]);
    }
  }

  // ---- normalize & store (row-major [B*T, N*H] bf16) ----
#pragma unroll
  for (int r = 0; r < 8; ++r) {
    const float inv = 1.0f / lrow[r];
    const int row = t0 + r + 8 * half;
    const size_t obase = ((size_t)(b * T + row)) * (NHEAD * H) + n * H;
#pragma unroll
    for (int nt = 0; nt < 8; ++nt) {
      V8FU u; u.v = acc[nt];
      O[obase + nt * 16 + l15] = (__bf16)(u.f[r] * inv);
    }
  }
}

// ---------------------------------------------------------------------------
// Host orchestration
// ---------------------------------------------------------------------------
extern "C" void kernel_launch(void* const* d_in, const int* in_sizes, int n_in,
                              void* d_out, int out_size, void* d_ws, size_t ws_size,
                              hipStream_t stream) {
  const float* x   = (const float*)d_in[0];
  const int* pos   = (const int*)d_in[1];
  const float* ln1 = (const float*)d_in[2];
  const float* qw  = (const float*)d_in[3];
  const float* qb  = (const float*)d_in[4];
  const float* kw  = (const float*)d_in[5];
  const float* kb  = (const float*)d_in[6];
  const float* vw  = (const float*)d_in[7];
  const float* vb  = (const float*)d_in[8];
  const float* ow  = (const float*)d_in[9];
  const float* ln2 = (const float*)d_in[10];
  const float* gw  = (const float*)d_in[11];
  const float* uw  = (const float*)d_in[12];
  const float* dw  = (const float*)d_in[13];
  float* out = (float*)d_out;

  constexpr int Btok = 4096, D = 2048, C = 2560, NH = 2048, I = 5504;

  char* base = (char*)d_ws;
  size_t off = 0;
  auto take = [&](size_t bytes) -> size_t {
    size_t cur = off;
    off += (bytes + 255) & ~(size_t)255;
    return cur;
  };
  const size_t o_hbf  = take((size_t)Btok * D * 2);       // h1 bf16
  const size_t o_wqkv = take((size_t)D * C * 2);          // packed qkv weight
  const size_t o_bias = take((size_t)C * 4);              // fused qkv bias f32
  const size_t o_qkv  = take((size_t)Btok * C * 4);       // qkv proj out f32
  const size_t o_qbf  = take((size_t)2 * 16 * 2048 * 128 * 2);  // q roped bf16
  const size_t o_kbf  = take((size_t)2 * 2 * 2048 * 128 * 2);   // k roped bf16
  const size_t o_vbf  = take((size_t)2 * 2 * 2048 * 128 * 2);   // v^T bf16
  const size_t o_attn = take((size_t)Btok * NH * 2);      // attention out bf16
  const size_t o_wo   = take((size_t)NH * D * 2);         // packed o_w
  const size_t o_wg   = take((size_t)D * I * 2);          // packed gate_w
  const size_t o_wu   = take((size_t)D * I * 2);          // packed up_w
  const size_t o_wd   = take((size_t)I * D * 2);          // packed down_w
  const size_t o_res  = take((size_t)Btok * D * 4);       // residual f32
  const size_t o_h2   = take((size_t)Btok * D * 2);       // h2 bf16
  // Aliases over dead regions:
  const size_t o_g    = 0;        // gate f32 [Btok x I] over region 1 (<= o_attn)
  const size_t o_fuse = o_attn;   // fused mlp bf16 over attn/w_o/w_gate (< o_wu)
  if (ws_size < off) return;

  __bf16* h_bf   = (__bf16*)(base + o_hbf);
  __bf16* w_qkv  = (__bf16*)(base + o_wqkv);
  float*  biasc  = (float*)(base + o_bias);
  float*  qkvf   = (float*)(base + o_qkv);
  __bf16* q_bf   = (__bf16*)(base + o_qbf);
  __bf16* k_bf   = (__bf16*)(base + o_kbf);
  __bf16* vT_bf  = (__bf16*)(base + o_vbf);
  __bf16* attn_b = (__bf16*)(base + o_attn);
  __bf16* w_o    = (__bf16*)(base + o_wo);
  __bf16* w_g    = (__bf16*)(base + o_wg);
  __bf16* w_u    = (__bf16*)(base + o_wu);
  __bf16* w_d    = (__bf16*)(base + o_wd);
  float*  res    = (float*)(base + o_res);
  __bf16* h2_bf  = (__bf16*)(base + o_h2);
  float*  g_f32  = (float*)(base + o_g);
  __bf16* fuse_b = (__bf16*)(base + o_fuse);

  // 1) rmsnorm(x) -> bf16
  rmsnorm_kernel<<<Btok, 256, 0, stream>>>(x, ln1, h_bf, D);

  // 2) pack all weights into WMMA B-fragment tile order (bf16)
  pack_qkv_w_kernel<<<(D * C) / 256, 256, 0, stream>>>(qw, kw, vw, w_qkv);
  pack_qkv_bias_kernel<<<10, 256, 0, stream>>>(qb, kb, vb, biasc);
  pack_b_kernel<<<(NH * D) / 256, 256, 0, stream>>>(ow, w_o, NH, D);
  pack_b_kernel<<<(D * I) / 256, 256, 0, stream>>>(gw, w_g, D, I);
  pack_b_kernel<<<(D * I) / 256, 256, 0, stream>>>(uw, w_u, D, I);
  pack_b_kernel<<<(I * D) / 256, 256, 0, stream>>>(dw, w_d, I, D);

  // 3) fused QKV projection (+bias) -> f32 [Btok x 2560]
  gemm_bf16_kernel<<<dim3(Btok / 128, C / 64), 256, 0, stream>>>(
      h_bf, w_qkv, biasc, nullptr, qkvf, nullptr, Btok, C, D, 1);

  // 4) RoPE(q)*H^-0.5, RoPE(k), transpose v -> bf16
  rope_q_kernel<<<(2 * 16 * 2048 * 64) / 256, 256, 0, stream>>>(qkvf, pos, q_bf);
  rope_k_kernel<<<(2 * 2 * 2048 * 64) / 256, 256, 0, stream>>>(qkvf, pos, k_bf);
  vtrans_kernel<<<(2 * 2 * 2048 * 128) / 256, 256, 0, stream>>>(qkvf, vT_bf);

  // 5) flash attention -> bf16 [Btok x N*H]
  attn_kernel<<<512, 256, 0, stream>>>(q_bf, k_bf, vT_bf, attn_b);

  // 6) o-projection + residual(x) -> res f32
  gemm_bf16_kernel<<<dim3(Btok / 128, D / 64), 256, 0, stream>>>(
      attn_b, w_o, nullptr, x, res, nullptr, Btok, D, NH, 2);

  // 7) rmsnorm(res) -> h2 bf16
  rmsnorm_kernel<<<Btok, 256, 0, stream>>>(res, ln2, h2_bf, D);

  // 8) gate projection -> f32
  gemm_bf16_kernel<<<dim3(Btok / 128, I / 64), 256, 0, stream>>>(
      h2_bf, w_g, nullptr, nullptr, g_f32, nullptr, Btok, I, D, 0);

  // 9) up projection; fuse silu(gate)*up -> bf16
  gemm_bf16_kernel<<<dim3(Btok / 128, I / 64), 256, 0, stream>>>(
      h2_bf, w_u, nullptr, g_f32, nullptr, fuse_b, Btok, I, D, 3);

  // 10) down projection + residual(res) -> d_out f32
  gemm_bf16_kernel<<<dim3(Btok / 128, D / 64), 256, 0, stream>>>(
      fuse_b, w_d, nullptr, res, out, nullptr, Btok, D, I, 2);
}
